// DiffPoolGNNMIL_75368086110728
// MI455X (gfx1250) — matrix operand order repaired and checked by reading, not compiled, not run on hardware
//
#include <hip/hip_runtime.h>
#include <hip/hip_bf16.h>

typedef __bf16 bf16;
typedef __attribute__((ext_vector_type(8)))  __bf16 v8bf;
typedef __attribute__((ext_vector_type(16))) __bf16 v16bf;
typedef __attribute__((ext_vector_type(8)))  float  v8f;

#define N_NODES 32000
#define NUM_GRAPHS 32
#define NODES_PER_GRAPH 1000
#define N_EDGES 256000
#define IN_DIM 1024
#define HIDDEN 256
#define NCLUST 8
// H column layout: [Hl1 (0..255) | Hla (256..263) | Hr1 (264..519) | Hra (520..527)]
#define NCOL 528          // 33 tiles of 16
#define NAGG 264          // columns that get edge-aggregated
#define NTILES_N 33
#define KSTEPS 32         // 1024 / 32

// ---------------------------------------------------------------------------
// 1) Split x (f32) into bf16 hi + bf16 lo residual (for 3-WMMA f32 emulation)
// ---------------------------------------------------------------------------
__global__ __launch_bounds__(256) void split_x_kernel(const float* __restrict__ x,
                                                      bf16* __restrict__ xh,
                                                      bf16* __restrict__ xl) {
    size_t i = (size_t)blockIdx.x * 256 + threadIdx.x;   // exactly 32768000 threads
    float v = x[i];
    bf16 h = (bf16)v;
    xh[i] = h;
    xl[i] = (bf16)(v - (float)h);
}

// ---------------------------------------------------------------------------
// 2) Pack concatenated weights [Wl1|Wla|Wr1|Wra] -> B-fragment order, split hi/lo.
//    Per (j,kstep): 32 lanes x 16 halves. ISA B layout (16-bit, 32x16):
//    lanes 0-15: n=lane, K=h<8?h:h+8 ; lanes 16-31: n=lane-16, K=h<8?h+8:h+16.
// ---------------------------------------------------------------------------
__global__ __launch_bounds__(256) void pack_w_kernel(const float* __restrict__ Wl1,
                                                     const float* __restrict__ Wr1,
                                                     const float* __restrict__ Wla,
                                                     const float* __restrict__ Wra,
                                                     bf16* __restrict__ wfh,
                                                     bf16* __restrict__ wfl) {
    int idx = blockIdx.x * 256 + threadIdx.x;            // 33*32*512 = 540672
    if (idx >= NTILES_N * KSTEPS * 512) return;
    int h    = idx & 15;
    int lane = (idx >> 4) & 31;
    int ks   = (idx >> 9) & 31;
    int j    = idx >> 14;
    int n = j * 16 + (lane & 15);
    int k = ks * 32 + (h & 7) + ((h >> 3) << 4) + ((lane >> 4) << 3);
    float w;
    if      (n < 256) w = Wl1[k * 256 + n];
    else if (n < 264) w = Wla[k * 8   + (n - 256)];
    else if (n < 520) w = Wr1[k * 256 + (n - 264)];
    else              w = Wra[k * 8   + (n - 520)];
    bf16 hi = (bf16)w;
    wfh[idx] = hi;
    wfl[idx] = (bf16)(w - (float)hi);
}

// ---------------------------------------------------------------------------
// 3) Fused GEMM: H[32000x528] = x[32000x1024] @ W[1024x528], split-bf16 3-WMMA.
//    8 waves/block; wave w owns M-tile (bx*8+w), all waves share N-tile by.
//    B fragments (hi+lo, 64 KB) staged in LDS once; A fragments from L2.
// ---------------------------------------------------------------------------
__global__ __launch_bounds__(256) void gemm_kernel(const bf16* __restrict__ xh,
                                                   const bf16* __restrict__ xl,
                                                   const bf16* __restrict__ wfh,
                                                   const bf16* __restrict__ wfl,
                                                   float* __restrict__ H) {
    __shared__ v16bf bh_sh[1024];   // [ks*32 + lane] -> 16 halves ; 32 KB
    __shared__ v16bf bl_sh[1024];   // 32 KB
    const int tid = threadIdx.x;
    const int j = blockIdx.y;
    {   // cooperative copy of this N-tile's B fragments (hi+lo): 2 x 2048 uint4
        const uint4* sH = (const uint4*)(wfh + (size_t)j * 16384);
        const uint4* sL = (const uint4*)(wfl + (size_t)j * 16384);
        uint4* dH = (uint4*)bh_sh;
        uint4* dL = (uint4*)bl_sh;
        #pragma unroll
        for (int c = tid; c < 2048; c += 256) { dH[c] = sH[c]; dL[c] = sL[c]; }
    }
    __syncthreads();

    const int wave = tid >> 5, lane = tid & 31;
    const int mtile = blockIdx.x * 8 + wave;
    const int m = mtile * 16 + (lane & 15);
    const int kb = (lane >> 4) * 8;   // ISA 16-bit A layout: lanes16-31 start at K=8
    const bf16* rowH = xh + (size_t)m * IN_DIM;
    const bf16* rowL = xl + (size_t)m * IN_DIM;

    v8f acc0 = {}; v8f acc1 = {}; v8f acc2 = {};   // independent -> no WMMA RAW chain
    #pragma unroll 4
    for (int ks = 0; ks < KSTEPS; ++ks) {
        const int k0 = ks * 32 + kb;
        // A fragment: halves 0-7 = K [k0..k0+7], halves 8-15 = K [k0+16..k0+23]
        v8bf a0 = *(const v8bf*)(rowH + k0);
        v8bf a1 = *(const v8bf*)(rowH + k0 + 16);
        v8bf l0 = *(const v8bf*)(rowL + k0);
        v8bf l1 = *(const v8bf*)(rowL + k0 + 16);
        v16bf Ah = __builtin_shufflevector(a0, a1, 0,1,2,3,4,5,6,7,8,9,10,11,12,13,14,15);
        v16bf Al = __builtin_shufflevector(l0, l1, 0,1,2,3,4,5,6,7,8,9,10,11,12,13,14,15);
        v16bf Bh = bh_sh[ks * 32 + lane];
        v16bf Bl = bl_sh[ks * 32 + lane];
        acc0 = __builtin_amdgcn_wmma_f32_16x16x32_bf16(false, Ah, false, Bh, (short)0, acc0, false, false);
        acc1 = __builtin_amdgcn_wmma_f32_16x16x32_bf16(false, Ah, false, Bl, (short)0, acc1, false, false);
        acc2 = __builtin_amdgcn_wmma_f32_16x16x32_bf16(false, Al, false, Bh, (short)0, acc2, false, false);
    }
    // C/D layout: VGPR r -> lanes 0-15: M=r ; lanes 16-31: M=8+r ; N=lane&15
    const int n  = j * 16 + (lane & 15);
    const int mb = mtile * 16 + ((lane >> 4) ? 8 : 0);
    #pragma unroll
    for (int r = 0; r < 8; ++r)
        H[(size_t)(mb + r) * NCOL + n] = acc0[r] + acc1[r] + acc2[r];
}

// ---------------------------------------------------------------------------
// 4) Edge mean-aggregation: Agg[dst, 0..263] += H[src, 0..263]; cnt[dst] += 1
// ---------------------------------------------------------------------------
__global__ __launch_bounds__(256) void edge_agg_kernel(const int* __restrict__ src,
                                                       const int* __restrict__ dst,
                                                       const float* __restrict__ H,
                                                       float* __restrict__ Agg,
                                                       float* __restrict__ cnt) {
    const int e = blockIdx.x;
    const int s = src[e], d = dst[e];
    const int t = threadIdx.x;
    atomicAdd(&Agg[(size_t)d * NAGG + t], H[(size_t)s * NCOL + t]);
    if (t < 8)
        atomicAdd(&Agg[(size_t)d * NAGG + 256 + t], H[(size_t)s * NCOL + 256 + t]);
    if (t == 0)
        atomicAdd(&cnt[d], 1.0f);
}

// ---------------------------------------------------------------------------
// 5) Per-node: Z=relu(.), S=softmax(.), accumulate Xp[g] += s (x) z, store Ss.
//    Block = 250 nodes of one graph; thread t owns hidden channel t.
// ---------------------------------------------------------------------------
__global__ __launch_bounds__(256) void node_finalize_kernel(const float* __restrict__ Agg,
                                                            const float* __restrict__ cnt,
                                                            const float* __restrict__ H,
                                                            const float* __restrict__ bl1,
                                                            const float* __restrict__ bla,
                                                            float* __restrict__ Ss,
                                                            float* __restrict__ Xp) {
    __shared__ float s_sh[8];
    const int g = blockIdx.x >> 2;
    const int base = g * NODES_PER_GRAPH + (blockIdx.x & 3) * 250;
    const int t = threadIdx.x;
    const float b1 = bl1[t];
    float xpacc[8] = {0.f, 0.f, 0.f, 0.f, 0.f, 0.f, 0.f, 0.f};

    for (int nn = 0; nn < 250; ++nn) {
        const int i = base + nn;
        const float c = cnt[i];
        float a = Agg[(size_t)i * NAGG + t];
        a = (c > 0.f) ? a / c : 0.f;
        float z = fmaxf(a + b1 + H[(size_t)i * NCOL + 264 + t], 0.f);
        if (t < 8) {
            float sa = Agg[(size_t)i * NAGG + 256 + t];
            sa = (c > 0.f) ? sa / c : 0.f;
            s_sh[t] = sa + bla[t] + H[(size_t)i * NCOL + 520 + t];
        }
        __syncthreads();
        if (t == 0) {
            float mx = s_sh[0];
            #pragma unroll
            for (int q = 1; q < 8; ++q) mx = fmaxf(mx, s_sh[q]);
            float ex[8], sum = 0.f;
            #pragma unroll
            for (int q = 0; q < 8; ++q) { ex[q] = expf(s_sh[q] - mx); sum += ex[q]; }
            #pragma unroll
            for (int q = 0; q < 8; ++q) s_sh[q] = ex[q] / sum;
        }
        __syncthreads();
        if (t < 8) Ss[(size_t)i * 8 + t] = s_sh[t];
        #pragma unroll
        for (int q = 0; q < 8; ++q) xpacc[q] += s_sh[q] * z;
        __syncthreads();   // protect s_sh before next iteration's write
    }
    #pragma unroll
    for (int q = 0; q < 8; ++q)
        atomicAdd(&Xp[(size_t)g * (NCLUST * HIDDEN) + q * HIDDEN + t], xpacc[q]);
}

// ---------------------------------------------------------------------------
// 6) Pooled adjacency: Ap[g] += Ss[src]^T (x) Ss[dst]  (LDS-staged ds_add_f32)
// ---------------------------------------------------------------------------
__global__ __launch_bounds__(256) void edge_ap_kernel(const int* __restrict__ src,
                                                      const int* __restrict__ dst,
                                                      const float* __restrict__ Ss,
                                                      float* __restrict__ Ap) {
    __shared__ float ap_sh[NUM_GRAPHS * 64];
    const int t = threadIdx.x;
    for (int c = t; c < NUM_GRAPHS * 64; c += 256) ap_sh[c] = 0.f;
    __syncthreads();
    const int e = blockIdx.x * 256 + t;       // exactly 256000 threads
    const int s = src[e], d = dst[e];
    const int g = s / NODES_PER_GRAPH;
    float ss[8], sd[8];
    #pragma unroll
    for (int q = 0; q < 8; ++q) { ss[q] = Ss[(size_t)s * 8 + q]; sd[q] = Ss[(size_t)d * 8 + q]; }
    #pragma unroll
    for (int ci = 0; ci < 8; ++ci)
        #pragma unroll
        for (int cj = 0; cj < 8; ++cj)
            atomicAdd(&ap_sh[g * 64 + ci * 8 + cj], ss[ci] * sd[cj]);
    __syncthreads();
    for (int c = t; c < NUM_GRAPHS * 64; c += 256)
        if (ap_sh[c] != 0.f) atomicAdd(&Ap[c], ap_sh[c]);
}

// ---------------------------------------------------------------------------
// 7) Per-graph head: mask/deg, pooled mean, Zp GEMMs, classifier MLP -> out[g]
// ---------------------------------------------------------------------------
__global__ __launch_bounds__(256) void final_head_kernel(const float* __restrict__ Xp,
                                                         const float* __restrict__ Ap,
                                                         const float* __restrict__ Wl2,
                                                         const float* __restrict__ bl2,
                                                         const float* __restrict__ Wr2,
                                                         const float* __restrict__ Wc1,
                                                         const float* __restrict__ bc1,
                                                         const float* __restrict__ Wc2,
                                                         const float* __restrict__ bc2,
                                                         float* __restrict__ out) {
    __shared__ float xp_sh[2048];
    __shared__ float agg_sh[2048];
    __shared__ float zp_sh[2048];
    __shared__ float mask_sh[64];
    __shared__ float deg_sh[8];
    __shared__ float red[256];
    const int g = blockIdx.x, t = threadIdx.x;
    for (int c = t; c < 2048; c += 256) xp_sh[c] = Xp[(size_t)g * 2048 + c];
    if (t < 64) mask_sh[t] = (Ap[(size_t)g * 64 + t] != 0.f) ? 1.f : 0.f;
    __syncthreads();
    if (t < 8) {
        float ds = 0.f;
        #pragma unroll
        for (int i2 = 0; i2 < 8; ++i2) ds += mask_sh[i2 * 8 + t];
        deg_sh[t] = ds;
    }
    __syncthreads();
    #pragma unroll
    for (int jj = 0; jj < 8; ++jj) {      // agg[j][t] = sum_i mask[i][j]*Xp[i][t] / deg
        float acc = 0.f;
        #pragma unroll
        for (int i2 = 0; i2 < 8; ++i2) acc += mask_sh[i2 * 8 + jj] * xp_sh[i2 * 256 + t];
        float dg = deg_sh[jj];
        agg_sh[jj * 256 + t] = (dg > 0.f) ? acc / dg : 0.f;
    }
    __syncthreads();
    for (int jj = 0; jj < 8; ++jj) {      // Zp = relu(agg@Wl2 + bl2 + Xp@Wr2)
        float acc = bl2[t];
        for (int k = 0; k < 256; ++k)
            acc += agg_sh[jj * 256 + k] * Wl2[k * 256 + t]
                 + xp_sh[jj * 256 + k]  * Wr2[k * 256 + t];
        zp_sh[jj * 256 + t] = fmaxf(acc, 0.f);
    }
    __syncthreads();
    float acc = bc1[t];                   // h = relu(g @ Wc1 + bc1)
    for (int k = 0; k < 2048; ++k) acc += zp_sh[k] * Wc1[k * 256 + t];
    red[t] = fmaxf(acc, 0.f) * Wc2[t];
    __syncthreads();
    for (int off = 128; off > 0; off >>= 1) {
        if (t < off) red[t] += red[t + off];
        __syncthreads();
    }
    if (t == 0) out[g] = red[0] + bc2[0];
}

// ---------------------------------------------------------------------------
extern "C" void kernel_launch(void* const* d_in, const int* in_sizes, int n_in,
                              void* d_out, int out_size, void* d_ws, size_t ws_size,
                              hipStream_t stream) {
    const float* x   = (const float*)d_in[0];
    const int*   ei  = (const int*)d_in[1];       // [2, E]: src then dst
    const float* Wl1 = (const float*)d_in[3];
    const float* bl1 = (const float*)d_in[4];
    const float* Wr1 = (const float*)d_in[5];
    const float* Wla = (const float*)d_in[6];
    const float* bla = (const float*)d_in[7];
    const float* Wra = (const float*)d_in[8];
    const float* Wl2 = (const float*)d_in[9];
    const float* bl2 = (const float*)d_in[10];
    const float* Wr2 = (const float*)d_in[11];
    const float* Wc1 = (const float*)d_in[12];
    const float* bc1 = (const float*)d_in[13];
    const float* Wc2 = (const float*)d_in[14];
    const float* bc2 = (const float*)d_in[15];
    const int* srcI = ei;
    const int* dstI = ei + N_EDGES;

    char* ws = (char*)d_ws;
    // Persistent region
    float* H   = (float*)(ws);                               // 32000*528*4 = 67,584,000
    bf16*  XH  = (bf16*)(ws + 67584000);                     // 65,536,000
    bf16*  XL  = (bf16*)(ws + 133120000);                    // 65,536,000
    bf16*  WFH = (bf16*)(ws + 198656000);                    // 1,081,344
    bf16*  WFL = (bf16*)(ws + 199737344);                    // 1,081,344 (end ~200.8 MB)
    // Alias region: overlays XH, used only AFTER the GEMM has consumed XH
    char*  al  = ws + 67584000;
    float* Agg = (float*)(al);                               // 32000*264*4 = 33,792,000
    float* Cnt = (float*)(al + 33792000);                    // 128,000
    float* Ssb = (float*)(al + 33920000);                    // 1,024,000
    float* Xp  = (float*)(al + 34944000);                    // 262,144
    float* Ap  = (float*)(al + 35206144);                    // 8,192  (end 35,214,336)

    split_x_kernel<<<128000, 256, 0, stream>>>(x, XH, XL);
    pack_w_kernel<<<2112, 256, 0, stream>>>(Wl1, Wr1, Wla, Wra, WFH, WFL);
    gemm_kernel<<<dim3(250, NTILES_N), 256, 0, stream>>>(XH, XL, WFH, WFL, H);
    // zero accumulators (after GEMM: this overwrites the XH alias region)
    hipMemsetAsync(al, 0, 35214336, stream);
    edge_agg_kernel<<<N_EDGES, 256, 0, stream>>>(srcI, dstI, H, Agg, Cnt);
    node_finalize_kernel<<<NUM_GRAPHS * 4, 256, 0, stream>>>(Agg, Cnt, H, bl1, bla, Ssb, Xp);
    edge_ap_kernel<<<N_EDGES / 256, 256, 0, stream>>>(srcI, dstI, Ssb, Ap);
    final_head_kernel<<<NUM_GRAPHS, 256, 0, stream>>>(Xp, Ap, Wl2, bl2, Wr2,
                                                      Wc1, bc1, Wc2, bc2, (float*)d_out);
}
